// GCNWithAggregator_Resnet_33878702031050
// MI455X (gfx1250) — matrix operand order repaired and checked by reading, not compiled, not run on hardware
//
#include <hip/hip_runtime.h>

// ---------------------------------------------------------------------------
// CDNA5 (gfx1250, wave32) BF16 WMMA GEMM + GCN aggregation pipeline.
// D = relu?(A@B + bias). A,B,C are FP32 in memory; operands are converted to
// BF16 and fed to v_wmma_f32_16x16x32_bf16 (FP32 accumulate). The A tile is
// staged to LDS by the Tensor Data Mover (double-buffered, TENSORcnt).
// ---------------------------------------------------------------------------

typedef __attribute__((ext_vector_type(16))) __bf16 v16bf;
typedef __attribute__((ext_vector_type(8)))  __bf16 v8bf;
typedef __attribute__((ext_vector_type(8)))  float  v8f;
typedef __attribute__((ext_vector_type(4)))  unsigned int u32x4;
typedef __attribute__((ext_vector_type(8)))  int i32x8;
typedef __attribute__((ext_vector_type(4)))  int i32x4;

#define BM 128          // rows per workgroup tile
#define BN 64           // cols per workgroup tile
#define BK 32           // k per step (matches 16x16x32 WMMA)
#define LDAF_S 36       // LDS row stride for A (f32): 144B = TDM pad 128B+16B
#define LDB_S 40        // LDS col stride for B (halves): 80B, bank-skewed

#if __has_builtin(__builtin_amdgcn_tensor_load_to_lds)
#define HAS_TDM 1
#else
#define HAS_TDM 0
#endif

#if HAS_TDM
// Issue one TDM 2D tile load: tile BK x BM f32 from gptr (row stride lda
// elements) into LDS at lds_addr, padding each 128B row with 16B (-> 36 f32
// LDS stride). OOB rows/cols (per rem_k/rem_m) are zero-filled by the TDM.
__device__ __forceinline__ void tdm_load_a_tile(const float* gptr,
                                                unsigned lds_addr,
                                                int rem_k, int rem_m, int lda)
{
  unsigned long long ga = (unsigned long long)(uintptr_t)gptr;
  u32x4 g0;
  g0[0] = 1u;                                           // count=1, user mode
  g0[1] = lds_addr;                                     // LDS byte address
  g0[2] = (unsigned)(ga & 0xFFFFFFFFu);                 // global_addr[31:0]
  g0[3] = (unsigned)((ga >> 32) & 0x1FFFFFFu)           // global_addr[56:32]
        | (2u << 30);                                   // type=2 (image/D#)
  unsigned tk = (unsigned)rem_k, tm = (unsigned)rem_m;
  i32x8 g1;
  g1[0] = (int)((2u << 16)        // data_size = 2 -> 4 bytes
              | (1u << 20)        // pad_enable
              | (4u << 22)        // pad_interval: 32 dwords (=128B) per row
              | (3u << 25));      // pad_amount:   4 dwords (=16B)
  g1[1] = (int)((tk & 0xFFFFu) << 16);                  // tensor_dim0 lo16
  g1[2] = (int)(((tk >> 16) & 0xFFFFu) | ((tm & 0xFFFFu) << 16)); // d0 hi|d1 lo
  g1[3] = (int)(((tm >> 16) & 0xFFFFu) | ((unsigned)BK << 16));   // d1 hi|tile0
  g1[4] = (int)(unsigned)BM;                            // tile_dim1 | tile_dim2=0
  g1[5] = lda;                                          // dim0_stride lo32
  g1[6] = 0;                                            // dim0_stride hi | dim1_stride lo
  g1[7] = 0;
  i32x4 z4 = {0, 0, 0, 0};
#if defined(__clang_major__) && (__clang_major__ >= 23)
  i32x8 z8 = {0, 0, 0, 0, 0, 0, 0, 0};
  __builtin_amdgcn_tensor_load_to_lds(g0, g1, z4, z4, z8, 0);
#else
  __builtin_amdgcn_tensor_load_to_lds(g0, g1, z4, z4, 0);
#endif
}
#endif

__global__ __launch_bounds__(256) void gemm_bf16_wmma(
    const float* __restrict__ A, const float* __restrict__ B,
    float* __restrict__ C, const float* __restrict__ bias,
    int M, int N, int K, int lda, int ldb, int ldc, int do_relu)
{
  __shared__ float  sAf[2][BM * LDAF_S];  // f32 A tiles (TDM destination)
  __shared__ __bf16 sB[BN * LDB_S];       // [col][k] transposed bf16

  const int tid  = threadIdx.x;
  const int lane = tid & 31;
  const int wave = tid >> 5;          // 0..7 : wave w owns rows [bm+16w, +16)
  const int bm = blockIdx.y * BM;
  const int bn = blockIdx.x * BN;
  const int l15 = lane & 15;
  const int lhi = lane >> 4;
  const int nk = K / BK;

  v8f acc[4] = {v8f{}, v8f{}, v8f{}, v8f{}};   // 16x64 per wave

#if HAS_TDM
  const unsigned ldsA0 = (unsigned)(uintptr_t)(void*)&sAf[0][0];
  const unsigned ldsA1 = (unsigned)(uintptr_t)(void*)&sAf[1][0];
  if (wave == 0)     // EXEC is ignored by TDM; one wave issues for the block
    tdm_load_a_tile(A + (size_t)bm * lda, ldsA0, K, M - bm, lda);
#endif

  for (int it = 0; it < nk; ++it) {
    const int cur = it & 1;
    const int k0  = it * BK;

#if HAS_TDM
    if (wave == 0) {
      if (it + 1 < nk) {
        // prefetch next A tile into the other buffer (freed by last barrier)
        tdm_load_a_tile(A + (size_t)bm * lda + (k0 + BK),
                        cur ? ldsA0 : ldsA1, K - (k0 + BK), M - bm, lda);
        __builtin_amdgcn_s_wait_tensorcnt(1);   // tile `it` has landed
      } else {
        __builtin_amdgcn_s_wait_tensorcnt(0);
      }
    }
#else
    // ---- manual stage A tile: 128x32 f32 (4 float4 per thread) ----
#pragma unroll
    for (int j = 0; j < 4; ++j) {
      int f   = tid + 256 * j;
      int row = f >> 3;
      int kq  = f & 7;
      float4 v = make_float4(0.f, 0.f, 0.f, 0.f);
      int gr = bm + row;
      if (gr < M) v = *(const float4*)(A + (size_t)gr * lda + k0 + kq * 4);
      *(float4*)(&sAf[cur][row * LDAF_S + kq * 4]) = v;
    }
#endif

    // ---- stage B tile: 32x64 f32 -> bf16, transposed to [n][k] ----
#pragma unroll
    for (int j = 0; j < 8; ++j) {
      int i  = tid + 256 * j;         // 0..2047
      int kk = i >> 6;                // 0..31
      int nn = i & 63;                // 0..63
      const float* gp = B + (size_t)(k0 + kk) * ldb + bn + nn;
      if (j == 0 && (k0 + BK) < K)
        __builtin_prefetch(gp + (size_t)BK * ldb, 0, 1);  // global_prefetch_b8
      sB[nn * LDB_S + kk] = (__bf16)(*gp);
    }
    __syncthreads();

    // ---- A fragment (ISA 7.12.2, 16-bit A 16x32):
    //   lanes 0-15:  row = l, K 0..7 and 16..23
    //   lanes 16-31: row = l-16, K 8..15 and 24..31
    const float* pa = &sAf[cur][(wave * 16 + l15) * LDAF_S + lhi * 8];
    float4 f0 = *(const float4*)(pa);
    float4 f1 = *(const float4*)(pa + 4);
    float4 f2 = *(const float4*)(pa + 16);
    float4 f3 = *(const float4*)(pa + 20);
    v16bf afrag;
    afrag[0]  = (__bf16)f0.x; afrag[1]  = (__bf16)f0.y;
    afrag[2]  = (__bf16)f0.z; afrag[3]  = (__bf16)f0.w;
    afrag[4]  = (__bf16)f1.x; afrag[5]  = (__bf16)f1.y;
    afrag[6]  = (__bf16)f1.z; afrag[7]  = (__bf16)f1.w;
    afrag[8]  = (__bf16)f2.x; afrag[9]  = (__bf16)f2.y;
    afrag[10] = (__bf16)f2.z; afrag[11] = (__bf16)f2.w;
    afrag[12] = (__bf16)f3.x; afrag[13] = (__bf16)f3.y;
    afrag[14] = (__bf16)f3.z; afrag[15] = (__bf16)f3.w;

#pragma unroll
    for (int t = 0; t < 4; ++t) {
      // B fragment: lane holds column n = t*16 + l15; K 0..7,16..23 (lanes<16) or
      // K 8..15,24..31 (lanes>=16), matching the A fragment.
      const __bf16* pb = sB + (t * 16 + l15) * LDB_S + lhi * 8;
      v8bf b0 = *(const v8bf*)(pb);
      v8bf b1 = *(const v8bf*)(pb + 16);
      v16bf bfrag;
#pragma unroll
      for (int i = 0; i < 8; ++i) { bfrag[i] = b0[i]; bfrag[8 + i] = b1[i]; }
      acc[t] = __builtin_amdgcn_wmma_f32_16x16x32_bf16(
          false, afrag, false, bfrag, (short)0, acc[t], false, false);
    }
    __syncthreads();
  }

  // ---- epilogue: C/D layout: vgpr v, lanes 0-15 -> M=v, lanes 16-31 -> M=8+v
  const int m0 = bm + wave * 16 + lhi * 8;
#pragma unroll
  for (int t = 0; t < 4; ++t) {
    int col = bn + t * 16 + l15;
    float bv = bias ? bias[col] : 0.0f;
#pragma unroll
    for (int v = 0; v < 8; ++v) {
      int row = m0 + v;
      if (row < M) {
        float o = acc[t][v] + bv;
        if (do_relu) o = fmaxf(o, 0.0f);
        C[(size_t)row * ldc + col] = o;
      }
    }
  }
}

// ---------------------------------------------------------------------------
// Graph / elementwise helper kernels
// ---------------------------------------------------------------------------

__global__ __launch_bounds__(256) void copy_gene_kernel(
    const float* __restrict__ x, float* __restrict__ xt,
    int n_gene, int in_d4, int goid_d4)   // widths in float4 units
{
  size_t tid = (size_t)blockIdx.x * blockDim.x + threadIdx.x;
  size_t total = (size_t)n_gene * in_d4;
  if (tid >= total) return;
  int row = (int)(tid / in_d4);
  int c4  = (int)(tid % in_d4);
  ((float4*)xt)[(size_t)row * in_d4 + c4] =
      ((const float4*)x)[(size_t)row * goid_d4 + c4];
}

__global__ __launch_bounds__(256) void deg_init_kernel(float* __restrict__ deg, int n) {
  int i = blockIdx.x * blockDim.x + threadIdx.x;
  if (i < n) deg[i] = 1.0f;                       // self loop
}

__global__ __launch_bounds__(256) void deg_count_kernel(
    const int* __restrict__ dst, float* __restrict__ deg, int E)
{
  int e = blockIdx.x * blockDim.x + threadIdx.x;
  if (e < E) atomicAdd(&deg[dst[e]], 1.0f);
}

__global__ __launch_bounds__(256) void dinv_kernel(
    const float* __restrict__ deg, float* __restrict__ dinv, int n)
{
  int i = blockIdx.x * blockDim.x + threadIdx.x;
  if (i < n) dinv[i] = rsqrtf(deg[i]);            // deg >= 1 always
}

// agg[i][c] = bias[c] + xw[i][c] * dinv[i]^2   (bias + self-loop message)
__global__ __launch_bounds__(256) void self_loop_bias_kernel(
    const float* __restrict__ xw, const float* __restrict__ dinv,
    const float* __restrict__ bias, float* __restrict__ agg, int n, int cshift)
{
  size_t tid = (size_t)blockIdx.x * blockDim.x + threadIdx.x;
  size_t total = (size_t)n << cshift;
  if (tid >= total) return;
  int i = (int)(tid >> cshift);
  int c = (int)(tid & ((1 << cshift) - 1));
  float di = dinv[i];
  agg[tid] = bias[c] + xw[tid] * di * di;
}

// agg[dst][c] += xw[src][c] * dinv[src]*dinv[dst]
__global__ __launch_bounds__(256) void edge_scatter_kernel(
    const float* __restrict__ xw, const int* __restrict__ src,
    const int* __restrict__ dst, const float* __restrict__ dinv,
    float* __restrict__ agg, int E, int cshift)
{
  size_t tid = (size_t)blockIdx.x * blockDim.x + threadIdx.x;
  size_t total = (size_t)E << cshift;
  if (tid >= total) return;
  int e = (int)(tid >> cshift);
  int c = (int)(tid & ((1 << cshift) - 1));
  int s = src[e], d = dst[e];
  float norm = dinv[s] * dinv[d];
  atomicAdd(agg + ((size_t)d << cshift) + c, xw[((size_t)s << cshift) + c] * norm);
}

// res <- relu(agg) + res
__global__ __launch_bounds__(256) void relu_add_kernel(
    const float* __restrict__ agg, float* __restrict__ res, size_t total)
{
  size_t tid = (size_t)blockIdx.x * blockDim.x + threadIdx.x;
  if (tid < total) res[tid] = fmaxf(agg[tid], 0.0f) + res[tid];
}

// out[i][:2] = [h2[i] | istj[i]] @ Wf + bf
__global__ __launch_bounds__(256) void fusion_kernel(
    const float* __restrict__ h2, const float* __restrict__ istj,
    const float* __restrict__ Wf, const float* __restrict__ bf_,
    float* __restrict__ out, int n_gene, int h2dim)
{
  int i = blockIdx.x * blockDim.x + threadIdx.x;
  if (i >= n_gene) return;
  float a0 = bf_[0], a1 = bf_[1];
  const float* r = h2 + (size_t)i * h2dim;
  for (int c = 0; c < h2dim; ++c) {
    float v = r[c];
    a0 += v * Wf[c * 2 + 0];
    a1 += v * Wf[c * 2 + 1];
  }
  float t = istj[i];
  a0 += t * Wf[h2dim * 2 + 0];
  a1 += t * Wf[h2dim * 2 + 1];
  out[(size_t)i * 2 + 0] = a0;
  out[(size_t)i * 2 + 1] = a1;
}

// ---------------------------------------------------------------------------
// Launch
// ---------------------------------------------------------------------------

static inline void launch_gemm(const float* A, const float* B, float* C,
                               const float* bias, int M, int N, int K,
                               int lda, int ldb, int ldc, int relu,
                               hipStream_t s)
{
  dim3 grid(N / BN, (M + BM - 1) / BM);
  gemm_bf16_wmma<<<grid, 256, 0, s>>>(A, B, C, bias, M, N, K, lda, ldb, ldc, relu);
}

static inline int grid1(size_t total) { return (int)((total + 255) / 256); }

extern "C" void kernel_launch(void* const* d_in, const int* in_sizes, int n_in,
                              void* d_out, int out_size, void* d_ws, size_t ws_size,
                              hipStream_t stream)
{
  const float* x    = (const float*)d_in[0];
  const float* istj = (const float*)d_in[1];
  const int*   src  = (const int*)d_in[2];
  const int*   dst  = (const int*)d_in[3];
  const float* Wg1  = (const float*)d_in[4];
  const float* bg1  = (const float*)d_in[5];
  const float* Wg2  = (const float*)d_in[6];
  const float* bg2  = (const float*)d_in[7];
  const float* W1   = (const float*)d_in[8];
  const float* b1   = (const float*)d_in[9];
  const float* W2   = (const float*)d_in[10];
  const float* b2   = (const float*)d_in[11];
  const float* Wr1  = (const float*)d_in[12];
  const float* br1  = (const float*)d_in[13];
  const float* Wr2  = (const float*)d_in[14];
  const float* br2  = (const float*)d_in[15];
  const float* Wf   = (const float*)d_in[16];
  const float* bff  = (const float*)d_in[17];
  float* out = (float*)d_out;

  // Shapes derived from flat sizes (no device-scalar reads; capture-safe).
  const int N_nodes = in_sizes[1];                 // istj: (N,1)
  const int E       = in_sizes[2];
  const int GOID_D  = in_sizes[0] / N_nodes;       // 4096
  const int HG1     = in_sizes[5];                 // bg1: 1024
  const int IN_D    = in_sizes[7];                 // bg2: 512
  const int H1      = in_sizes[9];                 // b1 : 256
  const int H2      = in_sizes[11];                // b2 : 128
  const int n_gene  = out_size / 2;                // out: (n_gene, 2)
  const int n_goid  = N_nodes - n_gene;
  int cs1 = 0; while ((1 << cs1) < H1) ++cs1;      // log2(H1)
  int cs2 = 0; while ((1 << cs2) < H2) ++cs2;      // log2(H2)

  // Workspace layout (bytes). buf_h (GEMM1 output, n_goid*HG1) overlays
  // bufA+bufB, which are first written only after buf_h is dead.
  char* ws = (char*)d_ws;
  size_t sz_xt = (size_t)N_nodes * IN_D * 4;
  size_t sz_A  = (size_t)N_nodes * H1 * 4;
  float* xt   = (float*)(ws);
  float* bufA = (float*)(ws + sz_xt);                   // xw1 -> xw2
  float* bufB = (float*)(ws + sz_xt + sz_A);            // r1 -> h1 -> agg2
  float* bufC = (float*)(ws + sz_xt + 2 * sz_A);        // agg1 -> r2 -> h2
  float* bufH = bufA;                                   // GOID hidden (overlay)
  float* deg  = (float*)(ws + sz_xt + 3 * sz_A);
  float* dinv = deg + N_nodes;
  (void)ws_size; (void)n_in;

  // --- GOID encoder: h = relu(goid_x @ Wg1 + bg1) ---
  launch_gemm(x + (size_t)n_gene * GOID_D, Wg1, bufH, bg1,
              n_goid, HG1, GOID_D, GOID_D, HG1, HG1, 1, stream);
  // --- goid_t = relu(h @ Wg2 + bg2), written into x_t[n_gene:] ---
  launch_gemm(bufH, Wg2, xt + (size_t)n_gene * IN_D, bg2,
              n_goid, IN_D, HG1, HG1, IN_D, IN_D, 1, stream);
  // --- x_t[:n_gene] = x[:n_gene, :IN_D] ---
  copy_gene_kernel<<<grid1((size_t)n_gene * (IN_D / 4)), 256, 0, stream>>>(
      x, xt, n_gene, IN_D / 4, GOID_D / 4);

  // --- degree / symmetric norm (self-loops included) ---
  deg_init_kernel<<<grid1(N_nodes), 256, 0, stream>>>(deg, N_nodes);
  deg_count_kernel<<<grid1(E), 256, 0, stream>>>(dst, deg, E);
  dinv_kernel<<<grid1(N_nodes), 256, 0, stream>>>(deg, dinv, N_nodes);

  // --- layer 1: xw1 = x_t@W1 ; r1 = x_t@Wr1 + br1 ---
  launch_gemm(xt, W1, bufA, nullptr, N_nodes, H1, IN_D, IN_D, H1, H1, 0, stream);
  launch_gemm(xt, Wr1, bufB, br1,    N_nodes, H1, IN_D, IN_D, H1, H1, 0, stream);
  self_loop_bias_kernel<<<grid1((size_t)N_nodes << cs1), 256, 0, stream>>>(
      bufA, dinv, b1, bufC, N_nodes, cs1);
  edge_scatter_kernel<<<grid1((size_t)E << cs1), 256, 0, stream>>>(
      bufA, src, dst, dinv, bufC, E, cs1);
  relu_add_kernel<<<grid1((size_t)N_nodes << cs1), 256, 0, stream>>>(
      bufC, bufB, (size_t)N_nodes << cs1);          // h1 in bufB

  // --- layer 2: xw2 = h1@W2 ; r2 = h1@Wr2 + br2 ---
  launch_gemm(bufB, W2, bufA, nullptr, N_nodes, H2, H1, H1, H2, H2, 0, stream);
  launch_gemm(bufB, Wr2, bufC, br2,    N_nodes, H2, H1, H1, H2, H2, 0, stream);
  self_loop_bias_kernel<<<grid1((size_t)N_nodes << cs2), 256, 0, stream>>>(
      bufA, dinv, b2, bufB, N_nodes, cs2);          // agg2 reuses bufB (h1 dead)
  edge_scatter_kernel<<<grid1((size_t)E << cs2), 256, 0, stream>>>(
      bufA, src, dst, dinv, bufB, E, cs2);
  relu_add_kernel<<<grid1((size_t)N_nodes << cs2), 256, 0, stream>>>(
      bufB, bufC, (size_t)N_nodes << cs2);          // h2 in bufC

  // --- fusion head ---
  fusion_kernel<<<grid1(n_gene), 256, 0, stream>>>(
      bufC, istj, Wf, bff, out, n_gene, H2);
}
